// GDN_7438883356899
// MI455X (gfx1250) — compile-verified, matrix operand order checked by reading, not run on hardware
//
#include <hip/hip_runtime.h>
#include <hip/hip_bf16.h>
#include <math.h>

// Problem constants (match reference)
#define Bb    16
#define Nn    8192
#define Ld    64
#define Dd    64
#define Kk    16
#define ROWS  (Bb * Nn)          // 131072 rows of the [B*N, D] matrices
#define NEGBIG (-3.0e38f)
#define BN_EPS 1e-5f
#define CH    64                 // edge chunk for aggregation
#define NT    (Nn / 16)          // 512 column tiles

typedef __attribute__((ext_vector_type(16))) __bf16 v16bf;
typedef __attribute__((ext_vector_type(8)))  float  v8f;

__device__ __forceinline__ v8f wmma_bf16(v16bf a, v16bf b, v8f c) {
  // D = A(16x32 bf16) x B(32x16 bf16) + C(16x16 f32)
  return __builtin_amdgcn_wmma_f32_16x16x32_bf16(false, a, false, b, (short)0, c,
                                                 false, false);
}

__device__ __forceinline__ void put4(v16bf& v, int base, float4 f) {
  v[base + 0] = (__bf16)f.x;
  v[base + 1] = (__bf16)f.y;
  v[base + 2] = (__bf16)f.z;
  v[base + 3] = (__bf16)f.w;
}

// ---------------------------------------------------------------------------
// init: zero counters / BN accumulators (ws is poisoned by harness)
// ---------------------------------------------------------------------------
__global__ void init_kernel(int* __restrict__ cnt, float* __restrict__ bnsum,
                            float* __restrict__ bnss) {
  int i = blockIdx.x * blockDim.x + threadIdx.x;
  if (i < Nn) cnt[i] = 0;
  if (i < Dd) { bnsum[i] = 0.f; bnss[i] = 0.f; }
}

// ---------------------------------------------------------------------------
// Per-node prep: inverse norm of embedding row, sei = e.att_i[D:], sej likewise
// ---------------------------------------------------------------------------
__global__ void node_prep(const float* __restrict__ emb,
                          const float* __restrict__ atti,
                          const float* __restrict__ attj,
                          float* __restrict__ invn, float* __restrict__ sei,
                          float* __restrict__ sej) {
  int n = blockIdx.x * blockDim.x + threadIdx.x;
  if (n >= Nn) return;
  const float* ep = emb + (size_t)n * Dd;
  float ss = 0.f, si = 0.f, sj = 0.f;
#pragma unroll 8
  for (int d = 0; d < Dd; ++d) {
    float v = ep[d];
    ss += v * v;
    si += v * atti[Dd + d];
    sj += v * attj[Dd + d];
  }
  invn[n] = rsqrtf(ss);
  sei[n] = si;
  sej[n] = sj;
}

// ---------------------------------------------------------------------------
// Pack normalized embeddings into the WMMA 16-bit operand layout (bf16).
// Per 16-node block ct, lane l holds node ct*16+(l&15); K chunk base
// kb=(l<16)?0:8; 32 bf16 per lane: [K=kb..kb+7, kb+16..kb+23] then +32.
// This layout serves as BOTH the A operand (row block) and B operand (column
// tile) of V_WMMA_F32_16X16X32_BF16.
// ---------------------------------------------------------------------------
__global__ void pack_emb(const float* __restrict__ emb,
                         const float* __restrict__ invn,
                         __bf16* __restrict__ ebf) {
  const int lane = threadIdx.x;
  const int ct = blockIdx.x;
  const int m = lane & 15;
  const int kb = (lane < 16) ? 0 : 8;
  const int g = ct * 16 + m;
  const float s = invn[g];
  const float* bp = emb + (size_t)g * Dd;
  __bf16* op = ebf + ((size_t)ct * 32 + lane) * 32;
#pragma unroll
  for (int j = 0; j < 8; ++j) {
    op[j]      = (__bf16)(bp[kb + j] * s);
    op[8 + j]  = (__bf16)(bp[kb + 16 + j] * s);
    op[16 + j] = (__bf16)(bp[32 + kb + j] * s);
    op[24 + j] = (__bf16)(bp[32 + kb + 16 + j] * s);
  }
}

// Same packing for W_lin (4 column tiles -> 8 KB)
__global__ void pack_w(const float* __restrict__ Wl, __bf16* __restrict__ wbf) {
  const int lane = threadIdx.x;
  const int t = blockIdx.x;
  const int m = lane & 15;
  const int kb = (lane < 16) ? 0 : 8;
  const int cn = t * 16 + m;
  __bf16* op = wbf + ((size_t)t * 32 + lane) * 32;
#pragma unroll
  for (int j = 0; j < 8; ++j) {
    op[j]      = (__bf16)Wl[(kb + j) * Dd + cn];
    op[8 + j]  = (__bf16)Wl[(kb + 16 + j) * Dd + cn];
    op[16 + j] = (__bf16)Wl[(32 + kb + j) * Dd + cn];
    op[24 + j] = (__bf16)Wl[(32 + kb + 16 + j) * Dd + cn];
  }
}

// ---------------------------------------------------------------------------
// h = x @ W_lin via WMMA bf16. One wave per 16-row tile; A from x via float4
// loads + packed cvt, B from pre-packed wbf (vector loads, no cvt).
// ---------------------------------------------------------------------------
__global__ void gemm_h(const float* __restrict__ x,
                       const __bf16* __restrict__ wbf, float* __restrict__ h) {
  const int lane = threadIdx.x;
  const int rowBase = blockIdx.x * 16;
  const int m = lane & 15;
  const int kb4 = (lane < 16) ? 0 : 2;  // kb/4
  const float4* xq = (const float4*)(x + (size_t)(rowBase + m) * Ld);

  v16bf a0, a1;
  {
    float4 f0 = xq[kb4 + 0], f1 = xq[kb4 + 1];    // K kb..kb+7
    float4 f2 = xq[kb4 + 4], f3 = xq[kb4 + 5];    // K kb+16..kb+23
    float4 f4 = xq[kb4 + 8], f5 = xq[kb4 + 9];    // +32
    float4 f6 = xq[kb4 + 12], f7 = xq[kb4 + 13];
    put4(a0, 0, f0);  put4(a0, 4, f1);  put4(a0, 8, f2);  put4(a0, 12, f3);
    put4(a1, 0, f4);  put4(a1, 4, f5);  put4(a1, 8, f6);  put4(a1, 12, f7);
  }

  const v16bf* wop = (const v16bf*)wbf;
  const int rhalf = (lane < 16) ? 0 : 8;
#pragma unroll
  for (int t = 0; t < 4; ++t) {
    v16bf b0 = wop[(t * 32 + lane) * 2 + 0];
    v16bf b1 = wop[(t * 32 + lane) * 2 + 1];
    v8f c = {};
    c = wmma_bf16(a0, b0, c);
    c = wmma_bf16(a1, b1, c);
#pragma unroll
    for (int i = 0; i < 8; ++i) {
      h[(size_t)(rowBase + i + rhalf) * Dd + t * 16 + m] = c[i];
    }
  }
}

// ---------------------------------------------------------------------------
// sxi = h.att_i[:D], sxj = h.att_j[:D] per row of h
// ---------------------------------------------------------------------------
__global__ void sx_kernel(const float* __restrict__ h,
                          const float* __restrict__ atti,
                          const float* __restrict__ attj,
                          float* __restrict__ sxi, float* __restrict__ sxj) {
  int r = blockIdx.x * blockDim.x + threadIdx.x;
  if (r >= ROWS) return;
  const float* hp = h + (size_t)r * Dd;
  float si = 0.f, sj = 0.f;
#pragma unroll 8
  for (int d = 0; d < Dd; ++d) {
    float v = hp[d];
    si += v * atti[d];
    sj += v * attj[d];
  }
  sxi[r] = si;
  sxj[r] = sj;
}

// ---------------------------------------------------------------------------
// kNN: cosine similarity via WMMA on pre-packed normalized bf16 operands.
// Inner loop per tile: 4 vector loads + 2 WMMAs + LDS stage + top-15 scan.
// Produces nbr[N,16] (slot 15 = self-loop).
// ---------------------------------------------------------------------------
__global__ void knn_kernel(const __bf16* __restrict__ ebf,
                           int* __restrict__ nbr) {
  const int lane = threadIdx.x;
  const int rowBlk = blockIdx.x;
  const int rowBase = rowBlk * 16;
  const int rowG = rowBase + (lane & 15);
  const int m = lane & 15;

  const v16bf* eop = (const v16bf*)ebf;
  const v16bf a0 = eop[(rowBlk * 32 + lane) * 2 + 0];
  const v16bf a1 = eop[(rowBlk * 32 + lane) * 2 + 1];

  float tv[15];
  int ti[15];
#pragma unroll
  for (int q = 0; q < 15; ++q) { tv[q] = NEGBIG; ti[q] = 0; }
  float minV = NEGBIG;
  int minP = 0;

  __shared__ float tile[256];
  const int rhalf = (lane < 16) ? 0 : 8;

  for (int ct = 0; ct < NT; ++ct) {
    if (ct + 8 < NT)
      __builtin_prefetch((const void*)&eop[((ct + 8) * 32 + lane) * 2], 0, 1);
    const v16bf b0 = eop[(ct * 32 + lane) * 2 + 0];
    const v16bf b1 = eop[(ct * 32 + lane) * 2 + 1];
    v8f c = {};
    c = wmma_bf16(a0, b0, c);
    c = wmma_bf16(a1, b1, c);
#pragma unroll
    for (int i = 0; i < 8; ++i) tile[(i + rhalf) * 16 + m] = c[i];
    __syncthreads();

    if (lane < 16) {
      const int colBase = ct * 16;
#pragma unroll 4
      for (int n = 0; n < 16; ++n) {
        const int col = colBase + n;
        if (col != rowG) {
          const float v = tile[lane * 16 + n];
          if (v > minV) {
            // predicated replace keeps tv/ti in registers (no scratch)
#pragma unroll
            for (int q = 0; q < 15; ++q)
              if (q == minP) { tv[q] = v; ti[q] = col; }
            minV = tv[0]; minP = 0;
#pragma unroll
            for (int q = 1; q < 15; ++q)
              if (tv[q] < minV) { minV = tv[q]; minP = q; }
          }
        }
      }
    }
    __syncthreads();
  }

  if (lane < 16) {
    int* np = nbr + (size_t)rowG * Kk;
#pragma unroll
    for (int q = 0; q < 15; ++q) np[q] = ti[q];
    np[15] = rowG;  // self loop
  }
}

// ---------------------------------------------------------------------------
// CSR build: in-degree counts, exclusive scan, fill incoming-edge lists
// ---------------------------------------------------------------------------
__global__ void count_kernel(const int* __restrict__ nbr, int* __restrict__ cnt) {
  int idx = blockIdx.x * blockDim.x + threadIdx.x;
  if (idx >= Nn * Kk) return;
  atomicAdd(&cnt[nbr[idx]], 1);
}

__global__ void scan_kernel(const int* __restrict__ cnt, int* __restrict__ offs,
                            int* __restrict__ curs) {
  __shared__ int part[256];
  __shared__ int ex[256];
  const int t = threadIdx.x;
  int s = 0;
  for (int i = 0; i < Nn / 256; ++i) s += cnt[t * (Nn / 256) + i];
  part[t] = s;
  __syncthreads();
  if (t == 0) {
    int run = 0;
    for (int q = 0; q < 256; ++q) { ex[q] = run; run += part[q]; }
  }
  __syncthreads();
  int run = ex[t];
  for (int i = 0; i < Nn / 256; ++i) {
    int idx = t * (Nn / 256) + i;
    offs[idx] = run;
    curs[idx] = run;
    run += cnt[idx];
  }
  if (t == 255) offs[Nn] = run;
}

__global__ void fill_kernel(const int* __restrict__ nbr, int* __restrict__ curs,
                            int* __restrict__ inl) {
  int idx = blockIdx.x * blockDim.x + threadIdx.x;
  if (idx >= Nn * Kk) return;
  const int src = idx >> 4;          // node owning this outgoing edge
  const int tgt = nbr[idx];
  const int pos = atomicAdd(&curs[tgt], 1);
  inl[pos] = src;
}

// ---------------------------------------------------------------------------
// Aggregation per target node t (gather formulation, no float scatter):
//   w_e(b) = exp(lrelu(sxi[b,t]+sei[t]+sxj[b,src]+sej[src])),
//   z[b,t,:] = sum_e w_e h[b,src,:] / sum_e w_e + bias ;  s_hat = z * e[t]
// 64 threads = feature lanes; softmax denominators via LDS ds_add_f32;
// weights staged per chunk in LDS. BN partial sums via 2 global atomics/thread.
// exp needs no max-subtraction: |alpha| is O(1), result identical.
// ---------------------------------------------------------------------------
__global__ void agg_kernel(const float* __restrict__ h,
                           const float* __restrict__ emb,
                           const float* __restrict__ sxi,
                           const float* __restrict__ sxj,
                           const float* __restrict__ sei,
                           const float* __restrict__ sej,
                           const float* __restrict__ bias,
                           const int* __restrict__ offs,
                           const int* __restrict__ inl,
                           float* __restrict__ shat,
                           float* __restrict__ bnsum, float* __restrict__ bnss) {
  const int t = blockIdx.x;
  const int tid = threadIdx.x;
  const int beg = offs[t];
  const int deg = offs[t + 1] - beg;

  __shared__ float at[Bb];
  __shared__ float wsum[Bb];
  __shared__ float invw[Bb];
  __shared__ int   sbuf[CH];
  __shared__ float wbuf[CH * Bb];

  if (tid < Bb) {
    at[tid] = sxi[tid * Nn + t] + sei[t];
    wsum[tid] = 0.f;
  }
  __syncthreads();

  // pass 1: softmax denominators
  for (int idx = tid; idx < deg * Bb; idx += 64) {
    const int e = idx >> 4;
    const int b = idx & 15;
    const int s = inl[beg + e];
    float a = at[b] + sxj[b * Nn + s] + sej[s];
    a = (a > 0.f) ? a : 0.2f * a;
    atomicAdd(&wsum[b], __expf(a));
  }
  __syncthreads();
  if (tid < Bb) invw[tid] = 1.f / wsum[tid];
  __syncthreads();

  float acc[Bb];
#pragma unroll
  for (int b = 0; b < Bb; ++b) acc[b] = 0.f;

  // pass 2: chunked weighted gather-accumulate
  for (int c0 = 0; c0 < deg; c0 += CH) {
    const int cn = (deg - c0 < CH) ? (deg - c0) : CH;
    for (int e = tid; e < cn; e += 64) sbuf[e] = inl[beg + c0 + e];
    __syncthreads();
    for (int idx = tid; idx < cn * Bb; idx += 64) {
      const int e = idx >> 4;
      const int b = idx & 15;
      const int s = sbuf[e];
      float a = at[b] + sxj[b * Nn + s] + sej[s];
      a = (a > 0.f) ? a : 0.2f * a;
      wbuf[idx] = __expf(a) * invw[b];
    }
    __syncthreads();
    for (int e = 0; e < cn; ++e) {
      const int s = sbuf[e];
#pragma unroll
      for (int b = 0; b < Bb; ++b)
        acc[b] += wbuf[e * Bb + b] * h[((size_t)b * Nn + s) * Dd + tid];
    }
    __syncthreads();
  }

  const float et = emb[(size_t)t * Dd + tid];
  const float bg = bias[tid];
  float ps = 0.f, ps2 = 0.f;
#pragma unroll
  for (int b = 0; b < Bb; ++b) {
    const float z = acc[b] + bg;
    const float sh = z * et;
    shat[((size_t)b * Nn + t) * Dd + tid] = sh;
    ps += sh;
    ps2 += sh * sh;
  }
  atomicAdd(&bnsum[tid], ps);
  atomicAdd(&bnss[tid], ps2);
}

// ---------------------------------------------------------------------------
// BN finalize: scale/shift per feature (training-mode batch stats, ddof=0)
// ---------------------------------------------------------------------------
__global__ void bn_finalize(const float* __restrict__ bnsum,
                            const float* __restrict__ bnss,
                            const float* __restrict__ gam,
                            const float* __restrict__ bet,
                            float* __restrict__ bnsc, float* __restrict__ bnsh) {
  const int d = threadIdx.x;
  if (d >= Dd) return;
  const float inv = 1.f / (float)ROWS;
  const float mean = bnsum[d] * inv;
  const float var = bnss[d] * inv - mean * mean;
  const float sc = gam[d] * rsqrtf(var + BN_EPS);
  bnsc[d] = sc;
  bnsh[d] = bet[d] - mean * sc;
}

// ---------------------------------------------------------------------------
// Output: out[r] = sum_d relu(s_hat*scale+shift) * out_W[d] + out_b
// ---------------------------------------------------------------------------
__global__ void out_kernel(const float* __restrict__ shat,
                           const float* __restrict__ bnsc,
                           const float* __restrict__ bnsh,
                           const float* __restrict__ oW,
                           const float* __restrict__ ob,
                           float* __restrict__ out) {
  int r = blockIdx.x * blockDim.x + threadIdx.x;
  if (r >= ROWS) return;
  const float* sp = shat + (size_t)r * Dd;
  float acc = ob[0];
#pragma unroll 8
  for (int d = 0; d < Dd; ++d) {
    float v = sp[d] * bnsc[d] + bnsh[d];
    v = (v > 0.f) ? v : 0.f;
    acc += v * oW[d];
  }
  out[r] = acc;
}

// ---------------------------------------------------------------------------
extern "C" void kernel_launch(void* const* d_in, const int* in_sizes, int n_in,
                              void* d_out, int out_size, void* d_ws, size_t ws_size,
                              hipStream_t stream) {
  (void)in_sizes; (void)n_in; (void)out_size; (void)ws_size;
  const float* x    = (const float*)d_in[0];
  const float* emb  = (const float*)d_in[1];
  const float* Wl   = (const float*)d_in[2];
  const float* atti = (const float*)d_in[3];
  const float* attj = (const float*)d_in[4];
  const float* bias = (const float*)d_in[5];
  const float* gam  = (const float*)d_in[6];
  const float* bet  = (const float*)d_in[7];
  const float* oW   = (const float*)d_in[8];
  const float* ob   = (const float*)d_in[9];
  float* out = (float*)d_out;

  // Workspace layout: wide-aligned arrays first (keeps 32B alignment for the
  // packed bf16 operand arrays), 4B-aligned tails last.
  char* w = (char*)d_ws;
  float*  h    = (float*)w;  w += (size_t)ROWS * Dd * 4;       // 32 MB
  float*  shat = (float*)w;  w += (size_t)ROWS * Dd * 4;       // 32 MB
  __bf16* ebf  = (__bf16*)w; w += (size_t)NT * 32 * 32 * 2;    // 1 MB packed emb
  __bf16* wbf  = (__bf16*)w; w += (size_t)4 * 32 * 32 * 2;     // 8 KB packed W
  float*  sxi  = (float*)w;  w += (size_t)ROWS * 4;
  float*  sxj  = (float*)w;  w += (size_t)ROWS * 4;
  float*  sei  = (float*)w;  w += Nn * 4;
  float*  sej  = (float*)w;  w += Nn * 4;
  float*  invn = (float*)w;  w += Nn * 4;
  int*    nbr  = (int*)w;    w += (size_t)Nn * Kk * 4;
  int*    cnt  = (int*)w;    w += Nn * 4;
  int*    curs = (int*)w;    w += Nn * 4;
  int*    inl  = (int*)w;    w += (size_t)Nn * Kk * 4;
  float*  bnsum = (float*)w; w += Dd * 4;
  float*  bnss  = (float*)w; w += Dd * 4;
  float*  bnsc  = (float*)w; w += Dd * 4;
  float*  bnsh  = (float*)w; w += Dd * 4;
  int*    offs  = (int*)w;   w += (Nn + 1) * 4;

  init_kernel<<<Nn / 256, 256, 0, stream>>>(cnt, bnsum, bnss);
  node_prep<<<Nn / 256, 256, 0, stream>>>(emb, atti, attj, invn, sei, sej);
  pack_emb<<<NT, 32, 0, stream>>>(emb, invn, ebf);
  pack_w<<<4, 32, 0, stream>>>(Wl, wbf);
  gemm_h<<<ROWS / 16, 32, 0, stream>>>(x, wbf, h);
  sx_kernel<<<ROWS / 256, 256, 0, stream>>>(h, atti, attj, sxi, sxj);
  knn_kernel<<<NT, 32, 0, stream>>>(ebf, nbr);
  count_kernel<<<(Nn * Kk) / 256, 256, 0, stream>>>(nbr, cnt);
  scan_kernel<<<1, 256, 0, stream>>>(cnt, offs, curs);
  fill_kernel<<<(Nn * Kk) / 256, 256, 0, stream>>>(nbr, curs, inl);
  agg_kernel<<<Nn, 64, 0, stream>>>(h, emb, sxi, sxj, sei, sej, bias, offs, inl,
                                    shat, bnsum, bnss);
  bn_finalize<<<1, 64, 0, stream>>>(bnsum, bnss, gam, bet, bnsc, bnsh);
  out_kernel<<<ROWS / 256, 256, 0, stream>>>(shat, bnsc, bnsh, oW, ob, out);
}
